// HermitianAttention_24902220382612
// MI455X (gfx1250) — matrix operand order repaired and checked by reading, compiled, not run on hardware
//
#include <hip/hip_runtime.h>

typedef __bf16 bf16_t;
typedef __attribute__((ext_vector_type(16))) __bf16 v16bf;
typedef __attribute__((ext_vector_type(8)))  float  v8f;

#define B_  2
#define S_  2048
#define D_  1024
#define H_  16
#define HD_ 64
#define M_  (B_ * S_)      // 4096 tokens
#define SCALE_ 0.125f      // HEAD_DIM^-0.5 = 1/8

// ---------------------------------------------------------------------------
// WMMA wrapper: D = A(16x32 bf16) * B(32x16 bf16) + C(16x16 f32)
// ---------------------------------------------------------------------------
__device__ __forceinline__ v8f wmma_bf16(v16bf a, v16bf b, v8f c) {
  return __builtin_amdgcn_wmma_f32_16x16x32_bf16(
      /*neg_a=*/false, a, /*neg_b=*/false, b,
      /*c_mod=*/(short)0, c, /*reuse_a=*/false, /*reuse_b=*/false);
}

// Negate a packed bf16 x16 vector by flipping sign bits (8 dword XORs).
__device__ __forceinline__ v16bf negbf(v16bf x) {
  union { v16bf v; unsigned u[8]; } t;
  t.v = x;
#pragma unroll
  for (int i = 0; i < 8; ++i) t.u[i] ^= 0x80008000u;
  return t.v;
}

// A tile 16x32 (MxK) from row-major storage, element (0,0) at p, row stride ld.
// ISA layout: lanes 0-15 (half=0): VGPR0-3 K=0..7, VGPR4-7 K=16..23;
//             lanes 16-31: +8.  -> two contiguous 8-elt (16B) loads per lane.
__device__ __forceinline__ v16bf load_a16x32(const bf16_t* p, int ld) {
  int lane = threadIdx.x & 31;
  int m = lane & 15, half = lane >> 4;
  const bf16_t* row = p + m * ld + 8 * half;
  v16bf a;
  ((uint4*)&a)[0] = *(const uint4*)(row);        // K = 0..7   (+8*half)
  ((uint4*)&a)[1] = *(const uint4*)(row + 16);   // K = 16..23 (+8*half)
  return a;
}

// B tile 32x16 (KxN) where element (k,n) = p[n*ld + k]  (k contiguous).
// lanes 0-15: K=0..7,16..23; lanes 16-31: +8 -> two 16B loads per lane.
__device__ __forceinline__ v16bf load_b_kcontig(const bf16_t* p, int ld) {
  int lane = threadIdx.x & 31;
  int n = lane & 15, half = lane >> 4;
  const bf16_t* q = p + n * ld + 8 * half;
  v16bf b;
  ((uint4*)&b)[0] = *(const uint4*)(q);
  ((uint4*)&b)[1] = *(const uint4*)(q + 16);
  return b;
}

// ---------------------------------------------------------------------------
// f32 -> bf16 conversion (elementwise, layout preserved) — for z
// ---------------------------------------------------------------------------
__global__ void __launch_bounds__(256)
k_cvt_bf16(const float* __restrict__ s, bf16_t* __restrict__ d, int n) {
  int i = blockIdx.x * blockDim.x + threadIdx.x;
  if (i < n) d[i] = (bf16_t)s[i];
}

// ---------------------------------------------------------------------------
// f32 [D_,D_] -> bf16 transposed (dst[n][k] = src[k][n]).
// LDS-tiled 32x32 transpose, block (32,8), grid (D_/32, D_/32).
// ---------------------------------------------------------------------------
__global__ void __launch_bounds__(256)
k_cvt_transpose(const float* __restrict__ src, bf16_t* __restrict__ dst) {
  __shared__ bf16_t tile[32][33];   // +1 pad: no bank conflicts
  int k0 = blockIdx.x * 32;
  int n0 = blockIdx.y * 32;
  int tx = threadIdx.x;             // 0..31
  int ty = threadIdx.y;             // 0..7
#pragma unroll
  for (int i = 0; i < 4; ++i)
    tile[ty + i * 8][tx] = (bf16_t)src[(size_t)(k0 + ty + i * 8) * D_ + (n0 + tx)];
  __syncthreads();
#pragma unroll
  for (int i = 0; i < 4; ++i)
    dst[(size_t)(n0 + ty + i * 8) * D_ + (k0 + tx)] = tile[tx][ty + i * 8];
}

// ---------------------------------------------------------------------------
// Complex projection: Cr = Ar*Wr - Ai*Wi, Ci = Ar*Wi + Ai*Wr
// A: [M_, D_] bf16 row-major; W given TRANSPOSED: WT[n][k] bf16 row-major.
// One wave -> 16x64 output strip (4 N-tiles), only 8 v8f accumulators:
//   Cr[t] += Ar*Wr + (-Ai)*Wi ; Ci[t] += Ar*Wi + Ai*Wr
// 16 WMMA per k-step, A reused 4x, all loads 16B/32B contiguous, no spills.
// vmode 0: write [B,H,S,hd] (Q,K).  vmode 1: write [B,H,hd,S] (V transposed).
// ---------------------------------------------------------------------------
__global__ void __launch_bounds__(256)
k_cproj(const bf16_t* __restrict__ Ar, const bf16_t* __restrict__ Ai,
        const bf16_t* __restrict__ WrT, const bf16_t* __restrict__ WiT,
        bf16_t* __restrict__ Or_, bf16_t* __restrict__ Oi_, int vmode) {
  int w  = blockIdx.x * (blockDim.x >> 5) + (threadIdx.x >> 5);
  int tm = w >> 4;          // 256 row tiles
  int tg = w & 15;          // 16 col groups of 64
  int lane = threadIdx.x & 31;
  int n = lane & 15, half = lane >> 4;

  v8f zero = {};
  v8f cr[4], ci[4];
#pragma unroll
  for (int t = 0; t < 4; ++t) { cr[t] = zero; ci[t] = zero; }

  const bf16_t* arow = Ar + (size_t)tm * 16 * D_;
  const bf16_t* irow = Ai + (size_t)tm * 16 * D_;
  for (int kk = 0; kk < D_; kk += 32) {
    v16bf a_r  = load_a16x32(arow + kk, D_);
    v16bf a_i  = load_a16x32(irow + kk, D_);
    v16bf a_in = negbf(a_i);
#pragma unroll
    for (int t = 0; t < 4; ++t) {
      size_t nb = (size_t)(tg * 64 + t * 16) * D_ + kk;
      v16bf w_r = load_b_kcontig(WrT + nb, D_);
      v16bf w_i = load_b_kcontig(WiT + nb, D_);
      cr[t] = wmma_bf16(a_r,  w_r, cr[t]);
      cr[t] = wmma_bf16(a_in, w_i, cr[t]);
      ci[t] = wmma_bf16(a_r,  w_i, ci[t]);
      ci[t] = wmma_bf16(a_i,  w_r, ci[t]);
    }
  }
#pragma unroll
  for (int t = 0; t < 4; ++t) {
#pragma unroll
    for (int r = 0; r < 8; ++r) {
      int m = tm * 16 + r + 8 * half;        // token row
      int c = tg * 64 + t * 16 + n;          // model col
      int bb = m >> 11, ss = m & (S_ - 1);
      int hh = c >> 6, dd = c & 63;
      size_t addr;
      if (vmode == 0) addr = (((size_t)bb * H_ + hh) * S_ + ss) * HD_ + dd;
      else            addr = (((size_t)bb * H_ + hh) * HD_ + dd) * S_ + ss;
      Or_[addr] = (bf16_t)cr[t][r];
      Oi_[addr] = (bf16_t)ci[t][r];
    }
  }
}

// ---------------------------------------------------------------------------
// Flash-style Hermitian attention. One wave = one 16-query tile of one head.
// Q,K: [B,H,S,hd] bf16.  V: [B,H,hd,S] bf16 (transposed).  Out: [B,S,D] bf16.
// grid = 512 blocks x 256 threads (4096 waves = 32 heads * 128 q-tiles).
// ---------------------------------------------------------------------------
__global__ void __launch_bounds__(256)
k_attn(const bf16_t* __restrict__ Qr, const bf16_t* __restrict__ Qi,
       const bf16_t* __restrict__ Kr, const bf16_t* __restrict__ Ki,
       const bf16_t* __restrict__ Vr, const bf16_t* __restrict__ Vi,
       bf16_t* __restrict__ Or_, bf16_t* __restrict__ Oi_) {
  __shared__ bf16_t lds[8][16 * 32];   // per-wave 1KB p-tile bounce buffer
  int wid = threadIdx.x >> 5;
  int w   = blockIdx.x * 8 + wid;
  int bh  = w >> 7;                    // b*H + h  (0..31)
  int qt  = w & 127;                   // q tile (0..127)
  int lane = threadIdx.x & 31;
  int n = lane & 15, half = lane >> 4;

  const bf16_t* qr0 = Qr + ((size_t)bh * S_ + qt * 16) * HD_;
  const bf16_t* qi0 = Qi + ((size_t)bh * S_ + qt * 16) * HD_;
  v16bf aqr[2], aqi[2];
  aqr[0] = load_a16x32(qr0, HD_);      aqr[1] = load_a16x32(qr0 + 32, HD_);
  aqi[0] = load_a16x32(qi0, HD_);      aqi[1] = load_a16x32(qi0 + 32, HD_);

  float mrun[8], lrun[8];
  v8f zero = {};
  v8f o_r[4], o_i[4];
#pragma unroll
  for (int r = 0; r < 8; ++r) { mrun[r] = -3.0e38f; lrun[r] = 0.f; }
#pragma unroll
  for (int t = 0; t < 4; ++t) { o_r[t] = zero; o_i[t] = zero; }

  const bf16_t* kr_h = Kr + (size_t)bh * S_ * HD_;
  const bf16_t* ki_h = Ki + (size_t)bh * S_ * HD_;
  const bf16_t* vr_h = Vr + (size_t)bh * HD_ * S_;
  const bf16_t* vi_h = Vi + (size_t)bh * HD_ * S_;

  for (int kb = 0; kb < S_; kb += 32) {
    // ---- scores: Re(Q K^H) = Qr.Kr + Qi.Ki, two 16-key column tiles ----
    v8f sc[2]; sc[0] = zero; sc[1] = zero;
#pragma unroll
    for (int j = 0; j < 2; ++j) {
      const bf16_t* kp_r = kr_h + (size_t)(kb + 16 * j) * HD_;
      const bf16_t* kp_i = ki_h + (size_t)(kb + 16 * j) * HD_;
#pragma unroll
      for (int kc = 0; kc < 2; ++kc) {            // hd = 64 -> two K=32 steps
        v16bf bkr = load_b_kcontig(kp_r + kc * 32, HD_);
        v16bf bki = load_b_kcontig(kp_i + kc * 32, HD_);
        sc[j] = wmma_bf16(aqr[kc], bkr, sc[j]);
        sc[j] = wmma_bf16(aqi[kc], bki, sc[j]);
      }
    }
    // ---- online softmax (per-row state; rows stripe across 16-lane halves) --
    float p0[8], p1[8];
#pragma unroll
    for (int r = 0; r < 8; ++r) {
      float s0 = sc[0][r] * SCALE_, s1 = sc[1][r] * SCALE_;
      float mx = fmaxf(s0, s1);
#pragma unroll
      for (int off = 8; off >= 1; off >>= 1)
        mx = fmaxf(mx, __shfl_xor(mx, off, 32));  // reduce within 16-lane half
      float mn   = fmaxf(mrun[r], mx);
      float corr = __expf(mrun[r] - mn);
      float e0 = __expf(s0 - mn), e1 = __expf(s1 - mn);
      float rs = e0 + e1;
#pragma unroll
      for (int off = 8; off >= 1; off >>= 1)
        rs += __shfl_xor(rs, off, 32);
      lrun[r] = lrun[r] * corr + rs;
      mrun[r] = mn;
#pragma unroll
      for (int t = 0; t < 4; ++t) { o_r[t][r] *= corr; o_i[t][r] *= corr; }
      p0[r] = e0; p1[r] = e1;
    }
    // ---- C-layout p -> LDS (16x32 row-major) -> A-layout reload ----
    bf16_t* my = lds[wid];
#pragma unroll
    for (int r = 0; r < 8; ++r) {
      int m = r + 8 * half;
      my[m * 32 + n]      = (bf16_t)p0[r];
      my[m * 32 + 16 + n] = (bf16_t)p1[r];
    }
    v16bf pa = load_a16x32(my, 32);   // wave-private; DScnt ordering suffices
    // ---- O += P @ V  (V stored [hd, S] -> k-contiguous B tiles) ----
#pragma unroll
    for (int t = 0; t < 4; ++t) {
      v16bf bvr = load_b_kcontig(vr_h + (size_t)(t * 16) * S_ + kb, S_);
      v16bf bvi = load_b_kcontig(vi_h + (size_t)(t * 16) * S_ + kb, S_);
      o_r[t] = wmma_bf16(pa, bvr, o_r[t]);
      o_i[t] = wmma_bf16(pa, bvi, o_i[t]);
    }
  }

  // ---- normalize and store with merged heads: [B,S,D] ----
  int bb = bh >> 4, hh = bh & 15;
#pragma unroll
  for (int r = 0; r < 8; ++r) {
    int m   = r + 8 * half;
    int tok = qt * 16 + m;
    float inv = 1.0f / lrun[r];
    size_t rowbase = ((size_t)bb * S_ + tok) * D_ + hh * HD_;
#pragma unroll
    for (int t = 0; t < 4; ++t) {
      Or_[rowbase + t * 16 + n] = (bf16_t)(o_r[t][r] * inv);
      Oi_[rowbase + t * 16 + n] = (bf16_t)(o_i[t][r] * inv);
    }
  }
}

// ---------------------------------------------------------------------------
// Output complex projection (transposed weights), f32 stores into d_out.
// Same 16x64-per-wave / 8-accumulator structure as k_cproj.
// ---------------------------------------------------------------------------
__global__ void __launch_bounds__(256)
k_cproj_out(const bf16_t* __restrict__ Ar, const bf16_t* __restrict__ Ai,
            const bf16_t* __restrict__ WrT, const bf16_t* __restrict__ WiT,
            float* __restrict__ Yr, float* __restrict__ Yi) {
  int w  = blockIdx.x * (blockDim.x >> 5) + (threadIdx.x >> 5);
  int tm = w >> 4;
  int tg = w & 15;
  int lane = threadIdx.x & 31;
  int n = lane & 15, half = lane >> 4;

  v8f zero = {};
  v8f cr[4], ci[4];
#pragma unroll
  for (int t = 0; t < 4; ++t) { cr[t] = zero; ci[t] = zero; }

  const bf16_t* arow = Ar + (size_t)tm * 16 * D_;
  const bf16_t* irow = Ai + (size_t)tm * 16 * D_;
  for (int kk = 0; kk < D_; kk += 32) {
    v16bf a_r  = load_a16x32(arow + kk, D_);
    v16bf a_i  = load_a16x32(irow + kk, D_);
    v16bf a_in = negbf(a_i);
#pragma unroll
    for (int t = 0; t < 4; ++t) {
      size_t nb = (size_t)(tg * 64 + t * 16) * D_ + kk;
      v16bf w_r = load_b_kcontig(WrT + nb, D_);
      v16bf w_i = load_b_kcontig(WiT + nb, D_);
      cr[t] = wmma_bf16(a_r,  w_r, cr[t]);
      cr[t] = wmma_bf16(a_in, w_i, cr[t]);
      ci[t] = wmma_bf16(a_r,  w_i, ci[t]);
      ci[t] = wmma_bf16(a_i,  w_r, ci[t]);
    }
  }
#pragma unroll
  for (int t = 0; t < 4; ++t) {
#pragma unroll
    for (int r = 0; r < 8; ++r) {
      size_t m = (size_t)tm * 16 + r + 8 * half;
      size_t c = (size_t)(tg * 64 + t * 16 + n);
      Yr[m * D_ + c] = cr[t][r];
      Yi[m * D_ + c] = ci[t][r];
    }
  }
}

// ---------------------------------------------------------------------------
// Host launch
// ---------------------------------------------------------------------------
extern "C" void kernel_launch(void* const* d_in, const int* in_sizes, int n_in,
                              void* d_out, int out_size, void* d_ws, size_t ws_size,
                              hipStream_t stream) {
  (void)in_sizes; (void)n_in; (void)out_size; (void)ws_size;

  const float* zr = (const float*)d_in[0];
  const float* zi = (const float*)d_in[1];
  const float* wf[8];
  for (int i = 0; i < 8; ++i) wf[i] = (const float*)d_in[2 + i];

  const size_t NZ = (size_t)M_ * D_;   // 4M elements (token tensors)
  const size_t NW = (size_t)D_ * D_;   // 1M elements (weights)

  bf16_t* p = (bf16_t*)d_ws;
  bf16_t* zr_b = p; p += NZ;
  bf16_t* zi_b = p; p += NZ;
  bf16_t* wb[8];                       // transposed bf16 weights [out,in]
  for (int i = 0; i < 8; ++i) { wb[i] = p; p += NW; }
  bf16_t* qr = p; p += NZ;  bf16_t* qi = p; p += NZ;
  bf16_t* kr = p; p += NZ;  bf16_t* ki = p; p += NZ;
  bf16_t* vr = p; p += NZ;  bf16_t* vi = p; p += NZ;
  bf16_t* ar = p; p += NZ;  bf16_t* ai = p; p += NZ;   // attention outputs

  // 1) bf16 conversions: z elementwise, weights transposed (LDS-tiled)
  k_cvt_bf16<<<(int)((NZ + 255) / 256), 256, 0, stream>>>(zr, zr_b, (int)NZ);
  k_cvt_bf16<<<(int)((NZ + 255) / 256), 256, 0, stream>>>(zi, zi_b, (int)NZ);
  {
    dim3 tg(D_ / 32, D_ / 32), tb(32, 8);
    for (int i = 0; i < 8; ++i)
      k_cvt_transpose<<<tg, tb, 0, stream>>>(wf[i], wb[i]);
  }

  // 2) complex QKV projections (256 m-tiles x 16 n-groups, 8 waves/block)
  const int projBlocks = (256 * 16) / 8;   // 512
  k_cproj<<<projBlocks, 256, 0, stream>>>(zr_b, zi_b, wb[0], wb[1], qr, qi, 0);
  k_cproj<<<projBlocks, 256, 0, stream>>>(zr_b, zi_b, wb[2], wb[3], kr, ki, 0);
  k_cproj<<<projBlocks, 256, 0, stream>>>(zr_b, zi_b, wb[4], wb[5], vr, vi, 1);

  // 3) flash-style Hermitian attention (4096 waves)
  k_attn<<<512, 256, 0, stream>>>(qr, qi, kr, ki, vr, vi, ar, ai);

  // 4) output complex projection, f32 into d_out (yr | yi)
  float* yr = (float*)d_out;
  float* yi = yr + NZ;
  k_cproj_out<<<projBlocks, 256, 0, stream>>>(ar, ai, wb[6], wb[7], yr, yi);
}
